// MogrifierLayer_26568667693302
// MI455X (gfx1250) — matrix operand order repaired and checked by reading, not compiled, not run on hardware
//
#include <hip/hip_runtime.h>
#include <math.h>

// ---------------------------------------------------------------------------
// MI455X (gfx1250) Mogrifier-LSTM layer.
// Heavy math on v_wmma_f32_16x16x32_bf16 (bf16 A/B, f32 accum), f32 in memory,
// native v_cvt_pk_bf16_f32 while staging tiles into double-buffered LDS with a
// software pipeline (global loads for tile k+1 overlap WMMA on tile k).
// B tile staged transposed via K-pair packing: one b32 LDS store per 2 elems.
// ---------------------------------------------------------------------------

typedef __attribute__((ext_vector_type(16))) __bf16 v16bf;
typedef __attribute__((ext_vector_type(8)))  __bf16 v8bf;
typedef __attribute__((ext_vector_type(2)))  __bf16 v2bf;
typedef __attribute__((ext_vector_type(8)))  float  v8f;

#define BM 128
#define BN 128
#define BK 32
#define NTHREADS 256
#define LDK 40   // padded LDS row stride (elements)

static __device__ inline __bf16 f2bf(float f) { return (__bf16)f; }  // native cvt

static __device__ inline v2bf pk2(float lo, float hi) {
    v2bf v; v[0] = f2bf(lo); v[1] = f2bf(hi); return v;   // -> v_cvt_pk_bf16_f32
}

// C[M,N] (+)= A[M,K] @ B[K,N] (+ bias[N]).  Row-major f32.
// M,N multiples of 128; K multiple of 32.
template<bool HAS_BIAS, bool ACCUM>
__global__ __launch_bounds__(NTHREADS)
void gemm_bf16_wmma(const float* __restrict__ A, const float* __restrict__ B,
                    float* __restrict__ C, const float* __restrict__ bias,
                    int M, int N, int K)
{
    __shared__ alignas(16) __bf16 sA[2][BM][LDK];   // [buf][m][k]
    __shared__ alignas(16) __bf16 sB[2][BN][LDK];   // [buf][n][k] (B transposed)

    const int tid = threadIdx.x;
    const int bm  = blockIdx.y * BM;
    const int bn  = blockIdx.x * BN;

    // A loader: 16 contiguous f32 (row ar, cols ac..ac+15 of the K-slice)
    const int ar = tid >> 1;
    const int ac = (tid & 1) * 16;
    // B loader: K-pair rows (bkp, bkp+1) x 8 columns (bnb..bnb+7)
    const int bkp = (tid >> 4) * 2;       // 0,2,..,30
    const int bnb = (tid & 15) * 8;       // 0,8,..,120

    const int wave  = tid >> 5;
    const int lane  = tid & 31;
    const int wRow  = wave >> 2;    // 0..1 -> M offset 64*wRow
    const int wCol  = wave & 3;     // 0..3 -> N offset 32*wCol
    const int lmod  = lane & 15;
    const int lhalf = lane >> 4;

    const float* aPtr = A + (size_t)(bm + ar) * K + ac;       // +BK per step
    const float* bPtr = B + (size_t)bkp * N + (bn + bnb);     // +BK*N per step

    v8f acc[4][2];
#pragma unroll
    for (int tm = 0; tm < 4; ++tm)
#pragma unroll
        for (int tn = 0; tn < 2; ++tn)
#pragma unroll
            for (int r = 0; r < 8; ++r) acc[tm][tn][r] = 0.0f;

    float4 ra[4];            // A: row ar, 16 floats
    float4 rb[4];            // B: rows bkp/bkp+1, 8 floats each

    auto loadTile = [&](int kt) {
        const float* ap = aPtr + (size_t)kt * BK;
        const float* bp = bPtr + (size_t)kt * BK * N;
#pragma unroll
        for (int i = 0; i < 4; ++i) ra[i] = *(const float4*)(ap + i * 4);
        rb[0] = *(const float4*)(bp);
        rb[1] = *(const float4*)(bp + 4);
        rb[2] = *(const float4*)(bp + N);
        rb[3] = *(const float4*)(bp + N + 4);
    };

    auto stageTile = [&](int buf) {
        __bf16* da = &sA[buf][ar][ac];
#pragma unroll
        for (int i = 0; i < 4; ++i) {
            da[i*4+0] = f2bf(ra[i].x); da[i*4+1] = f2bf(ra[i].y);
            da[i*4+2] = f2bf(ra[i].z); da[i*4+3] = f2bf(ra[i].w);
        }
        // transposed B: (n, kpair) -> single packed b32 store
        const float r0[8] = { rb[0].x, rb[0].y, rb[0].z, rb[0].w,
                              rb[1].x, rb[1].y, rb[1].z, rb[1].w };
        const float r1[8] = { rb[2].x, rb[2].y, rb[2].z, rb[2].w,
                              rb[3].x, rb[3].y, rb[3].z, rb[3].w };
#pragma unroll
        for (int j = 0; j < 8; ++j)
            *(v2bf*)&sB[buf][bnb + j][bkp] = pk2(r0[j], r1[j]);
    };

    // ---- prologue: fetch tile 0 and stage into buffer 0 ----
    loadTile(0);
    stageTile(0);
    __syncthreads();

    const int nk = K / BK;
    for (int kt = 0; kt < nk; ++kt) {
        const int cur = kt & 1;
        const int nxt = cur ^ 1;
        const bool hasNext = (kt + 1) < nk;

        // ---- issue global loads for tile kt+1 (overlap with WMMA below) ----
        if (hasNext) loadTile(kt + 1);

        // ---- fragments per ISA 16-bit layouts, 8 WMMAs ----
        v16bf afrag[4], bfrag[2];
#pragma unroll
        for (int tm = 0; tm < 4; ++tm) {
            // A 16x32: lanes 0-15 -> K 0..7 | 16..23 ; lanes 16-31 -> K 8..15 | 24..31
            const __bf16* p = &sA[cur][wRow * 64 + tm * 16 + lmod][lhalf * 8];
            v8bf lo = *(const v8bf*)p;
            v8bf hi = *(const v8bf*)(p + 16);
            afrag[tm] = __builtin_shufflevector(lo, hi,
                0,1,2,3,4,5,6,7,8,9,10,11,12,13,14,15);
        }
#pragma unroll
        for (int tn = 0; tn < 2; ++tn) {
            // B 32x16: lanes 0-15 -> K 0..7 | 16..23 ; lanes 16-31 -> K 8..15 | 24..31 ; N = lane%16
            const __bf16* p = &sB[cur][wCol * 32 + tn * 16 + lmod][lhalf * 8];
            v8bf lo = *(const v8bf*)p;
            v8bf hi = *(const v8bf*)(p + 16);
            bfrag[tn] = __builtin_shufflevector(lo, hi,
                0,1,2,3,4,5,6,7,8,9,10,11,12,13,14,15);
        }
#pragma unroll
        for (int tm = 0; tm < 4; ++tm)
#pragma unroll
            for (int tn = 0; tn < 2; ++tn)
                acc[tm][tn] = __builtin_amdgcn_wmma_f32_16x16x32_bf16(
                    false, afrag[tm], false, bfrag[tn],
                    (short)0, acc[tm][tn], false, false);

        // ---- convert + stage tile kt+1 into the other buffer ----
        if (hasNext) stageTile(nxt);
        __syncthreads();
    }

    // ---- epilogue: D layout -> M = r + 8*lhalf (+tiles), N = lane%16 (+tiles) ----
#pragma unroll
    for (int tm = 0; tm < 4; ++tm) {
#pragma unroll
        for (int tn = 0; tn < 2; ++tn) {
            const int n = bn + wCol * 32 + tn * 16 + lmod;
            float bv = 0.0f;
            if (HAS_BIAS) bv = bias[n];
#pragma unroll
            for (int r = 0; r < 8; ++r) {
                const int m = bm + wRow * 64 + tm * 16 + lhalf * 8 + r;
                const size_t idx = (size_t)m * N + n;
                const float v = acc[tm][tn][r] + bv;
                if (ACCUM) C[idx] += v; else C[idx] = v;
            }
        }
    }
}

// ---------------- elementwise kernels ----------------
static __device__ inline float sigf(float x) { return 1.0f / (1.0f + expf(-x)); }

__global__ void ew_sig2mul(const float* __restrict__ u, const float* __restrict__ v,
                           float* __restrict__ out, int n) {
    int i = blockIdx.x * blockDim.x + threadIdx.x;
    if (i < n) out[i] = 2.0f * sigf(u[i]) * v[i];
}

__global__ void ew_act(float* __restrict__ buf, int mode, int n) {
    int i = blockIdx.x * blockDim.x + threadIdx.x;
    if (i < n) buf[i] = mode ? tanhf(buf[i]) : sigf(buf[i]);
}

__global__ void ew_cell(const float* __restrict__ f, const float* __restrict__ cc,
                        const float* __restrict__ ig, const float* __restrict__ j,
                        float* __restrict__ out, int n) {
    int i = blockIdx.x * blockDim.x + threadIdx.x;
    if (i < n) out[i] = f[i] * cc[i] + ig[i] * j[i];
}

__global__ void ew_hcell(const float* __restrict__ o, const float* __restrict__ c,
                         float* __restrict__ out, int n) {
    int i = blockIdx.x * blockDim.x + threadIdx.x;
    if (i < n) out[i] = o[i] * tanhf(c[i]);
}

// ---------------- host orchestration ----------------
static inline void gemm(hipStream_t s, const float* A, const float* B, float* C,
                        const float* bias, int M, int N, int K, int accum) {
    dim3 grid(N / BN, M / BM), block(NTHREADS);
    if (accum)
        hipLaunchKernelGGL((gemm_bf16_wmma<false, true>), grid, block, 0, s, A, B, C, bias, M, N, K);
    else if (bias)
        hipLaunchKernelGGL((gemm_bf16_wmma<true, false>), grid, block, 0, s, A, B, C, bias, M, N, K);
    else
        hipLaunchKernelGGL((gemm_bf16_wmma<false, false>), grid, block, 0, s, A, B, C, bias, M, N, K);
}

extern "C" void kernel_launch(void* const* d_in, const int* in_sizes, int n_in,
                              void* d_out, int out_size, void* d_ws, size_t ws_size,
                              hipStream_t stream) {
    (void)in_sizes; (void)n_in; (void)out_size; (void)ws_size;
    enum { Bd = 1024, E = 4096, DH = 1024, DW = 1024, DQ = 1024, R = 5 };

    const float* x   = (const float*)d_in[0];
    const float* wx  = (const float*)d_in[1];
    const float* wh  = (const float*)d_in[2];
    const float* bg  = (const float*)d_in[3];
    const float* dxW = (const float*)d_in[4];
    const float* dxb = (const float*)d_in[5];
    const float* dhW = (const float*)d_in[6];
    const float* dhb = (const float*)d_in[7];
    const float* qk  = (const float*)d_in[8];
    const float* qkW = (const float*)d_in[9];
    const float* qkb = (const float*)d_in[10];
    const float* hcW = (const float*)d_in[11];
    const float* hcb = (const float*)d_in[12];
    const float* heW = (const float*)d_in[13];
    const float* heb = (const float*)d_in[14];
    const float* ccW = (const float*)d_in[15];
    const float* ccb = (const float*)d_in[16];
    const float* ceW = (const float*)d_in[17];
    const float* ceb = (const float*)d_in[18];
    const float* h0  = (const float*)d_in[19];
    const float* c0  = (const float*)d_in[20];

    float* out   = (float*)d_out;
    float* o_out = out;                          // (B,E)
    float* h_out = out + (size_t)Bd * E;         // (B,DH)
    float* c_out = h_out + (size_t)Bd * DH;      // (B,DH)

    // workspace carve-up (f32), ~108 MiB total
    float* ws    = (float*)d_ws;
    float* x_cur = ws;  ws += (size_t)Bd * E;
    float* hbuf  = ws;  ws += (size_t)Bd * DH;
    float* gbuf  = ws;  ws += (size_t)Bd * Bd;
    float* gbuf2 = ws;  ws += (size_t)Bd * Bd;
    float* ubuf  = ws;  ws += (size_t)Bd * E;
    float* hcbuf = ws;  ws += (size_t)Bd * E;
    float* fbuf  = ws;  ws += (size_t)Bd * E;
    float* ibuf  = ws;  ws += (size_t)Bd * E;
    float* jbuf  = ws;  ws += (size_t)Bd * E;

    const int nBE = Bd * E;
    const dim3 ewb(256), ewg((nBE + 255) / 256);

    // ---- Mogrifier rounds ----
    const float* x_ptr = x;
    const float* h_ptr = h0;
    for (int i = 0; i < R; ++i) {
        // g = qk[i] @ qkW[i] + qkb[i]      (B,B)
        gemm(stream, qk + (size_t)i * Bd * DQ, qkW + (size_t)i * DQ * Bd,
             gbuf, qkb + (size_t)i * Bd, Bd, Bd, DQ, 0);
        if ((i & 1) == 0) {
            gemm(stream, gbuf, x_ptr, ubuf, nullptr, Bd, E, Bd, 0);       // u = g @ x
            gemm(stream, h_ptr, hcW, hcbuf, hcb, Bd, E, DH, 0);           // hc = h@hcW+hcb
            hipLaunchKernelGGL(ew_sig2mul, ewg, ewb, 0, stream, ubuf, hcbuf, ubuf, nBE);
            gemm(stream, ubuf, heW, hbuf, heb, Bd, DH, E, 0);             // h = s@heW+heb
            h_ptr = hbuf;
        } else {
            gemm(stream, h_ptr, hcW, hcbuf, hcb, Bd, E, DH, 0);           // hc
            gemm(stream, gbuf, hcbuf, ubuf, nullptr, Bd, E, Bd, 0);       // u = g @ hc
            hipLaunchKernelGGL(ew_sig2mul, ewg, ewb, 0, stream, ubuf, x_ptr, x_cur, nBE);
            x_ptr = x_cur;
        }
    }

    // hc = dense(h, hcW, hcb)
    gemm(stream, h_ptr, hcW, hcbuf, hcb, Bd, E, DH, 0);

    // ---- four LSTM gates ----
    float* gate_dst[4] = { fbuf, ibuf, jbuf, o_out };
    for (int k = 0; k < 4; ++k) {
        float* gt = gate_dst[k];
        gemm(stream, wx + (size_t)k * Bd * DW, dxW + (size_t)k * DW * Bd,
             gbuf, dxb + (size_t)k * Bd, Bd, Bd, DW, 0);                  // Gx (B,B)
        gemm(stream, wh + (size_t)k * Bd * DW, dhW + (size_t)k * DW * Bd,
             gbuf2, dhb + (size_t)k * Bd, Bd, Bd, DW, 0);                 // Gh (B,B)
        gemm(stream, gbuf, x_ptr, gt, bg + (size_t)k * E, Bd, E, Bd, 0);  // pre = Gx@x + b
        gemm(stream, gbuf2, hcbuf, gt, nullptr, Bd, E, Bd, 1);            // pre += Gh@hc
        hipLaunchKernelGGL(ew_act, ewg, ewb, 0, stream, gt, (k == 2) ? 1 : 0, nBE);
    }

    // cc = dense(c0, ccW, ccb)
    gemm(stream, c0, ccW, ubuf, ccb, Bd, E, DH, 0);
    // c = f*cc + i*j   (into fbuf)
    hipLaunchKernelGGL(ew_cell, ewg, ewb, 0, stream, fbuf, ubuf, ibuf, jbuf, fbuf, nBE);
    // h_cell = o*tanh(c)  (into ibuf)
    hipLaunchKernelGGL(ew_hcell, ewg, ewb, 0, stream, o_out, fbuf, ibuf, nBE);
    // h_new = dense(h_cell, heW, heb)
    gemm(stream, ibuf, heW, h_out, heb, Bd, DH, E, 0);
    // c_new = dense(c, ceW, ceb)
    gemm(stream, fbuf, ceW, c_out, ceb, Bd, DH, E, 0);
}
